// Attention_12360915878747
// MI455X (gfx1250) — matrix operand order repaired and checked by reading, not compiled, not run on hardware
//
#include <hip/hip_runtime.h>
#include <hip/hip_bf16.h>

typedef unsigned short u16;
typedef unsigned int   u32;
typedef __attribute__((ext_vector_type(16))) __bf16 v16bf;
typedef __attribute__((ext_vector_type(8)))  float  v8f;

#define HIDDEN   2048
#define NHEADS   16
#define HDIM     128
#define WINDOW   1024
#define SEQ      4096

union BF16x16 { v16bf v; u16 u[16]; u32 w[8]; uint4 q[2]; };

__device__ __forceinline__ u16 f2bf(float f) {
  __bf16 h = (__bf16)f;
  return __builtin_bit_cast(u16, h);
}
__device__ __forceinline__ float bf2f(u16 s) {
  union { u32 u; float f; } x; x.u = ((u32)s) << 16; return x.f;
}
__device__ __forceinline__ u32 pack2(float lo, float hi) {
  return (u32)f2bf(lo) | ((u32)f2bf(hi) << 16);
}

// load 16 bf16 as two 8-half groups separated by `gap` elements (compile-time)
__device__ __forceinline__ v16bf ld16x2(const u16* p, int gap) {
  BF16x16 t;
  t.q[0] = *(const uint4*)(p);
  t.q[1] = *(const uint4*)(p + gap);
  return t.v;
}

__device__ __forceinline__ v8f wmma_bf16(v16bf a, v16bf b, v8f c) {
  return __builtin_amdgcn_wmma_f32_16x16x32_bf16(false, a, false, b, (short)0, c,
                                                 false, false);
}

// ---------------- prologue kernels ----------------
__global__ void cast_bf16_kernel(const float4* __restrict__ x, uint2* __restrict__ y,
                                 int n4) {
  int i = blockIdx.x * blockDim.x + threadIdx.x;
  if (i < n4) {
    float4 v = x[i];
    uint2 o;
    o.x = pack2(v.x, v.y);
    o.y = pack2(v.z, v.w);
    y[i] = o;
  }
}

// wt[n][k] = bf16(w[k][n]), HIDDENxHIDDEN, tiled via LDS
__global__ void tcast_kernel(const float* __restrict__ w, u16* __restrict__ wt) {
  __shared__ float tile[32][33];
  int bx = blockIdx.x * 32, by = blockIdx.y * 32;
  for (int j = threadIdx.y; j < 32; j += 8)
    tile[j][threadIdx.x] = w[(size_t)(by + j) * HIDDEN + bx + threadIdx.x];
  __syncthreads();
  for (int j = threadIdx.y; j < 32; j += 8)
    wt[(size_t)(bx + j) * HIDDEN + by + threadIdx.x] = f2bf(tile[threadIdx.x][j]);
}

// in-place RoPE on bf16 [SEQ][NHEADS*HDIM]; `mul` folds the attention scale into Q
__global__ void rope_kernel(u16* __restrict__ X, float mul) {
  int idx = blockIdx.x * blockDim.x + threadIdx.x;  // SEQ*NHEADS*64
  if (idx >= SEQ * NHEADS * 64) return;
  int i = idx & 63;
  int h = (idx >> 6) & (NHEADS - 1);
  int l = idx >> 10;
  float inv = __expf(-(float)i * (9.210340371976184f / 64.0f));  // 10000^(-i/64)
  float f = (float)l * inv;
  float s, c;
  __sincosf(f, &s, &c);
  size_t base = (size_t)l * HIDDEN + h * HDIM + i;
  float x1 = bf2f(X[base]), x2 = bf2f(X[base + 64]);
  X[base]      = f2bf((x1 * c - x2 * s) * mul);
  X[base + 64] = f2bf((x2 * c + x1 * s) * mul);
}

// ---------------- WMMA GEMM: C[M,N] = A[M,K] * B, B given as Bt[N][K] ----------------
// MODE 0: bf16 C[M][N]; MODE 1: bf16 transposed Ct[N][M]; MODE 2: f32 C[M][N]
template <int MODE>
__global__ void __launch_bounds__(128, 1)
gemm16_kernel(const u16* __restrict__ A, const u16* __restrict__ Bt,
              void* __restrict__ Cout, int M, int N, int K) {
  int lane = threadIdx.x & 31;
  int wave = threadIdx.x >> 5;  // 0..3
  int r15 = lane & 15, sel = lane >> 4;
  int m0 = blockIdx.y * 64 + wave * 16;
  int n0 = blockIdx.x * 64;
  const u16* aP = A + (size_t)(m0 + r15) * K + sel * 16;
  const u16* bP = Bt + (size_t)(n0 + r15) * K + sel * 16;
  v8f acc[4] = {};
  for (int k0 = 0; k0 < K; k0 += 64) {
    if (k0 + 64 < K) {
      __builtin_prefetch(aP + k0 + 64, 0, 1);
      __builtin_prefetch(bP + k0 + 64, 0, 1);
    }
    // stage operands first: loads issue as one clause, waits stagger
    v16bf a0 = ld16x2(aP + k0, 8);
    v16bf a1 = ld16x2(aP + k0 + 32, 8);
    v16bf b0[4], b1[4];
#pragma unroll
    for (int j = 0; j < 4; ++j) b0[j] = ld16x2(bP + k0 + j * (16 * HIDDEN), 8);
#pragma unroll
    for (int j = 0; j < 4; ++j) b1[j] = ld16x2(bP + k0 + 32 + j * (16 * HIDDEN), 8);
#pragma unroll
    for (int j = 0; j < 4; ++j) acc[j] = wmma_bf16(a0, b0[j], acc[j]);
#pragma unroll
    for (int j = 0; j < 4; ++j) acc[j] = wmma_bf16(a1, b1[j], acc[j]);
  }
#pragma unroll
  for (int j = 0; j < 4; ++j) {
    if (MODE == 1) {
#pragma unroll
      for (int rr = 0; rr < 8; rr += 2) {
        int m = m0 + rr + 8 * sel;  // even
        int n = n0 + j * 16 + r15;
        *(u32*)((u16*)Cout + (size_t)n * M + m) = pack2(acc[j][rr], acc[j][rr + 1]);
      }
    } else {
#pragma unroll
      for (int rr = 0; rr < 8; ++rr) {
        int m = m0 + rr + 8 * sel;
        int n = n0 + j * 16 + r15;
        if (MODE == 0) ((u16*)Cout)[(size_t)m * N + n] = f2bf(acc[j][rr]);
        else           ((float*)Cout)[(size_t)m * N + n] = acc[j][rr];
      }
    }
  }
}

// ---------------- flash sliding-window attention (transposed-S scheme) ----------------
// grid (SEQ/16, NHEADS), block 32 (one wave). Q pre-scaled by 1/sqrt(HDIM).
// S^T = K*Q^T (A = K rows, B = Q): lane owns query = lane&15.
// O accumulated transposed: O^T = V^T * P^T (A = Vt rows) -> per-query stats in-lane.
__global__ void __launch_bounds__(32, 1)
attn_kernel(const u16* __restrict__ Q, const u16* __restrict__ Kc,
            const u16* __restrict__ Vt, u16* __restrict__ O) {
  int lane = threadIdx.x & 31;
  int r15 = lane & 15, sel = lane >> 4;
  int q0 = blockIdx.x * 16;
  int h = blockIdx.y;

  const u16* qBase = Q + (size_t)(q0 + r15) * HIDDEN + h * HDIM + sel * 8;
  v16bf qb[4];
#pragma unroll
  for (int kc = 0; kc < 4; ++kc) qb[kc] = ld16x2(qBase + kc * 32, 16);

  int kstart = q0 - (WINDOW - 1);
  if (kstart < 0) kstart = 0;
  kstart &= ~31;

  const u16* kBase = Kc + (size_t)(kstart + r15) * HIDDEN + h * HDIM + sel * 8;
  const u16* vBase = Vt + (size_t)(h * HDIM + r15) * SEQ + kstart + sel * 16;

  v8f acc[8] = {};
  float mq = -1e30f, lq = 0.f;
  int myq = q0 + r15;

  for (int kb = kstart; kb <= q0 + 15; kb += 32, kBase += 32 * HIDDEN, vBase += 32) {
    // stage all 8 K A-operand tiles (one clause), then S^T WMMAs
    v16bf ka[8];
#pragma unroll
    for (int kc = 0; kc < 4; ++kc) {
      ka[kc]     = ld16x2(kBase + kc * 32, 16);
      ka[4 + kc] = ld16x2(kBase + 16 * HIDDEN + kc * 32, 16);
    }
    v8f st0 = {}, st1 = {};
#pragma unroll
    for (int kc = 0; kc < 4; ++kc) {
      st0 = wmma_bf16(ka[kc],     qb[kc], st0);
      st1 = wmma_bf16(ka[4 + kc], qb[kc], st1);
    }
    // issue V loads now; softmax VALU below hides their latency
    v16bf va[8];
#pragma unroll
    for (int t = 0; t < 8; ++t) va[t] = ld16x2(vBase + t * (16 * SEQ), 8);

    // wave-uniform fast path: chunk fully inside every query's window
    bool full = (kb + 31 <= q0) && (kb > q0 + 15 - WINDOW);
    float v0[8], v1[8];
    float mx = -3.0e38f;
#pragma unroll
    for (int rr = 0; rr < 8; ++rr) {
      float a = st0[rr], b = st1[rr];
      if (!full) {
        int key0 = kb + rr + 8 * sel, key1 = key0 + 16;
        if (!(key0 <= myq && key0 > myq - WINDOW)) a = -3.0e38f;
        if (!(key1 <= myq && key1 > myq - WINDOW)) b = -3.0e38f;
      }
      v0[rr] = a; v1[rr] = b;
      mx = fmaxf(mx, fmaxf(a, b));
    }
    mx = fmaxf(mx, __shfl_xor(mx, 16));  // single cross-lane max
    float mnew = fmaxf(mq, mx);
    float sc = __expf(mq - mnew);
    float p0[8], p1[8], sum = 0.f;
#pragma unroll
    for (int rr = 0; rr < 8; ++rr) {
      p0[rr] = __expf(v0[rr] - mnew);
      p1[rr] = __expf(v1[rr] - mnew);
      sum += p0[rr] + p1[rr];
    }
    sum += __shfl_xor(sum, 16);          // single cross-lane sum
    lq = lq * sc + sum;
    mq = mnew;
#pragma unroll
    for (int t = 0; t < 8; ++t) acc[t] *= sc;

    // build P^T B-operand: keys kb+sel*16+0..15 for column myq (one xor-16 exchange)
    u32 b0[4], b1[4];
#pragma unroll
    for (int j = 0; j < 4; ++j) {
      b0[j] = pack2(p0[2 * j], p0[2 * j + 1]);
      b1[j] = pack2(p1[2 * j], p1[2 * j + 1]);
    }
    BF16x16 pt;
#pragma unroll
    for (int j = 0; j < 4; ++j) {
      u32 snd = sel ? b0[j] : b1[j];
      u32 rcv = __shfl_xor(snd, 16);
      pt.w[j]     = sel ? rcv : b0[j];
      pt.w[4 + j] = sel ? b1[j] : rcv;
    }
    // O^T += V^T * P^T
#pragma unroll
    for (int t = 0; t < 8; ++t) acc[t] = wmma_bf16(va[t], pt.v, acc[t]);
  }
  // epilogue: lane holds query myq (col), dims rr+8sel+16t (rows)
  float inv = 1.0f / lq;
  u16* orow = O + (size_t)myq * HIDDEN + h * HDIM + 8 * sel;
#pragma unroll
  for (int t = 0; t < 8; ++t)
#pragma unroll
    for (int rr = 0; rr < 8; rr += 2)
      *(u32*)(orow + t * 16 + rr) = pack2(acc[t][rr] * inv, acc[t][rr + 1] * inv);
}

extern "C" void kernel_launch(void* const* d_in, const int* in_sizes, int n_in,
                              void* d_out, int out_size, void* d_ws, size_t ws_size,
                              hipStream_t stream) {
  const float* hs = (const float*)d_in[0];
  const float* wq = (const float*)d_in[1];
  const float* wk = (const float*)d_in[2];
  const float* wv = (const float*)d_in[3];
  const float* wo = (const float*)d_in[4];
  float* out = (float*)d_out;

  const size_t SZ_ACT = (size_t)SEQ * HIDDEN * sizeof(u16);     // 16 MB
  const size_t SZ_W   = (size_t)HIDDEN * HIDDEN * sizeof(u16);  // 8 MB
  char* p = (char*)d_ws;
  u16* hsb = (u16*)p; p += SZ_ACT;
  u16* wqt = (u16*)p; p += SZ_W;
  u16* wkt = (u16*)p; p += SZ_W;
  u16* wvt = (u16*)p; p += SZ_W;
  u16* wot = (u16*)p; p += SZ_W;
  u16* Qb  = (u16*)p; p += SZ_ACT;
  u16* Kb  = (u16*)p; p += SZ_ACT;
  u16* VtB = (u16*)p; p += SZ_ACT;   // [HIDDEN][SEQ]
  u16* Ob  = (u16*)p; p += SZ_ACT;

  // 1) casts / transposes
  {
    int n4 = SEQ * HIDDEN / 4;
    cast_bf16_kernel<<<(n4 + 255) / 256, 256, 0, stream>>>((const float4*)hs,
                                                           (uint2*)hsb, n4);
    dim3 tg(HIDDEN / 32, HIDDEN / 32), tb(32, 8);
    tcast_kernel<<<tg, tb, 0, stream>>>(wq, wqt);
    tcast_kernel<<<tg, tb, 0, stream>>>(wk, wkt);
    tcast_kernel<<<tg, tb, 0, stream>>>(wv, wvt);
    tcast_kernel<<<tg, tb, 0, stream>>>(wo, wot);
  }
  // 2) projections (WMMA): Q, K normal; V stored transposed
  {
    dim3 grid(HIDDEN / 64, SEQ / 64);
    gemm16_kernel<0><<<grid, 128, 0, stream>>>(hsb, wqt, Qb, SEQ, HIDDEN, HIDDEN);
    gemm16_kernel<0><<<grid, 128, 0, stream>>>(hsb, wkt, Kb, SEQ, HIDDEN, HIDDEN);
    gemm16_kernel<1><<<grid, 128, 0, stream>>>(hsb, wvt, VtB, SEQ, HIDDEN, HIDDEN);
  }
  // 3) RoPE on Q (folds 1/sqrt(HDIM)) and K
  {
    int n = SEQ * NHEADS * 64;
    rope_kernel<<<(n + 255) / 256, 256, 0, stream>>>(Qb, 0.08838834764831845f);
    rope_kernel<<<(n + 255) / 256, 256, 0, stream>>>(Kb, 1.0f);
  }
  // 4) sliding-window flash attention (WMMA, transposed-S)
  attn_kernel<<<dim3(SEQ / 16, NHEADS), 32, 0, stream>>>(Qb, Kb, VtB, Ob);
  // 5) output projection (WMMA, f32 out)
  gemm16_kernel<2><<<dim3(HIDDEN / 64, SEQ / 64), 128, 0, stream>>>(Ob, wot, out,
                                                                   SEQ, HIDDEN, HIDDEN);
  (void)in_sizes; (void)n_in; (void)out_size; (void)ws_size;
}
